// RouterGuidedNegativeMiner_47450798686773
// MI455X (gfx1250) — matrix operand order repaired and checked by reading, not compiled, not run on hardware
//
#include <hip/hip_runtime.h>
#include <hip/hip_bf16.h>

#define EPSV 1e-8f

typedef __attribute__((ext_vector_type(16))) __bf16 v16bf;
typedef __attribute__((ext_vector_type(8)))  float  v8f;

constexpr int Bn   = 1024;
constexpr int Kn   = 512;
constexpr int En   = 64;
constexpr int TOPK = 64;
constexpr int Dn   = 769;

// ln(2) to convert v_log_f32 (log2) to natural log
#define LN2 0.69314718055994530942f

__device__ __forceinline__ float fast_ln(float x) {
    return LN2 * __builtin_amdgcn_logf(x);
}

// ---------------------------------------------------------------------------
// Kernel 1: per-b anchor normalization + C[b], then WMMA-based scoring:
//   score[b,k] = sum_e a[b,e] * ln(n_norm[b,k,e]+eps) - C[b]
// One block per b (256 threads = 8 waves). Each wave scores 16 k-rows per
// slab via two v_wmma_f32_16x16x32_bf16 (E=64 split into two K=32 chunks).
// ---------------------------------------------------------------------------
__global__ __launch_bounds__(256) void rg_score_kernel(
    const float* __restrict__ anchor,   // [B,E]
    const float* __restrict__ neg,      // [B,K,E]
    float* __restrict__ scores)         // [B,K]  (workspace)
{
    __shared__ float  sh_a[En];
    __shared__ __bf16 sh_abf[En];
    __shared__ float  sh_c[En];
    __shared__ float  sh_sum, sh_C;

    const int b   = blockIdx.x;
    const int tid = threadIdx.x;

    // ---- Phase 1: normalize anchor row, compute C = sum a*ln(a+eps) ----
    if (tid < En) sh_a[tid] = anchor[b * En + tid] + EPSV;
    __syncthreads();
    if (tid == 0) {
        float s = 0.f;
        for (int e = 0; e < En; ++e) s += sh_a[e];
        sh_sum = s;
    }
    __syncthreads();
    if (tid < En) {
        float a = sh_a[tid] * __builtin_amdgcn_rcpf(sh_sum);
        sh_abf[tid] = (__bf16)a;
        sh_c[tid]   = a * fast_ln(a + EPSV);
    }
    __syncthreads();
    if (tid == 0) {
        float s = 0.f;
        for (int e = 0; e < En; ++e) s += sh_c[e];
        sh_C = s;
    }
    __syncthreads();
    const float Cb = sh_C;

    const int wave = tid >> 5;
    const int lane = tid & 31;
    const int half = lane >> 4;     // which 16-lane half
    const int mrow = lane & 15;     // row within the 16x16 tile

    // ---- B fragments: anchor vector replicated into all 16 columns ----
    // B 32x16 bf16 layout: lanes 0-15 hold K={0..7,16..23}, lanes 16-31 K={8..15,24..31}
    v16bf bf0, bf1;
#pragma unroll
    for (int i = 0; i < 16; ++i) {
        bf0[i] = sh_abf[(i & 8) + half * 8 + i];        // e-chunk 0..31
        bf1[i] = sh_abf[32 + (i & 8) + half * 8 + i];   // e-chunk 32..63
    }

    // ---- Phase 2: score 128 k's per slab (8 waves x 16 rows), 4 slabs ----
    for (int slab = 0; slab < 4; ++slab) {
        const int kbase = slab * 128 + wave * 16;
        const int row   = kbase + mrow;
        const float* nrow = neg + ((size_t)b * Kn + row) * En;

        // A 16x32 bf16 layout: lane<16 owns K in {0..7,16..23}, lane>=16 owns
        // {8..15,24..31}. With two e-chunks this lane owns 4 contiguous
        // groups of 8 floats starting at {0,16,32,48} + half*8.
        float g[4][8];
#pragma unroll
        for (int gr = 0; gr < 4; ++gr) {
            const int start = gr * 16 + half * 8;
            const float4 lo = *(const float4*)(nrow + start);
            const float4 hi = *(const float4*)(nrow + start + 4);
            g[gr][0] = lo.x; g[gr][1] = lo.y; g[gr][2] = lo.z; g[gr][3] = lo.w;
            g[gr][4] = hi.x; g[gr][5] = hi.y; g[gr][6] = hi.z; g[gr][7] = hi.w;
        }

        // row sum of (n+eps): own 32 values + partner half-lane via shfl_xor
        float part = 0.f;
#pragma unroll
        for (int gr = 0; gr < 4; ++gr)
#pragma unroll
            for (int i = 0; i < 8; ++i) part += g[gr][i];
        const float rowsum = part + __shfl_xor(part, 16, 32) + En * EPSV;
        const float invS   = __builtin_amdgcn_rcpf(rowsum);

        // A fragments: ln((n+eps)*invS + eps) in bf16
        v16bf af0, af1;
#pragma unroll
        for (int i = 0; i < 8; ++i) {
            af0[i]     = (__bf16)fast_ln((g[0][i] + EPSV) * invS + EPSV);
            af0[8 + i] = (__bf16)fast_ln((g[1][i] + EPSV) * invS + EPSV);
            af1[i]     = (__bf16)fast_ln((g[2][i] + EPSV) * invS + EPSV);
            af1[8 + i] = (__bf16)fast_ln((g[3][i] + EPSV) * invS + EPSV);
        }

        v8f acc = {};
        acc = __builtin_amdgcn_wmma_f32_16x16x32_bf16(
            false, af0, false, bf0, (short)0, acc, false, false);
        acc = __builtin_amdgcn_wmma_f32_16x16x32_bf16(
            false, af1, false, bf1, (short)0, acc, false, false);

        // D layout: lane 0 holds M=0..7 (N=0), lane 16 holds M=8..15 (N=0)
        if ((lane & 15) == 0) {
#pragma unroll
            for (int r = 0; r < 8; ++r)
                scores[(size_t)b * Kn + kbase + half * 8 + r] = acc[r] - Cb;
        }
    }
}

// ---------------------------------------------------------------------------
// Kernel 2: per-b bitonic sort (descending; ties -> lower index first, like
// jax.lax.top_k), then emit top-64 scores and gather candidate rows.
// ---------------------------------------------------------------------------
__global__ __launch_bounds__(256) void rg_topk_gather_kernel(
    const float* __restrict__ scores,   // [B,K]
    const float* __restrict__ cand,     // [B,K,D]
    float* __restrict__ hard,           // [B,TOPK,D]
    float* __restrict__ out_scores)     // [B,TOPK]
{
    __shared__ float s[Kn];
    __shared__ int   sidx[Kn];

    const int b   = blockIdx.x;
    const int tid = threadIdx.x;

    for (int i = tid; i < Kn; i += 256) {
        s[i]    = scores[(size_t)b * Kn + i];
        sidx[i] = i;
    }
    __syncthreads();

    for (int ksz = 2; ksz <= Kn; ksz <<= 1) {
        for (int j = ksz >> 1; j > 0; j >>= 1) {
            for (int i = tid; i < Kn; i += 256) {
                const int p = i ^ j;
                if (p > i) {
                    const bool dir = ((i & ksz) == 0);   // descending region
                    const float si = s[i], sp = s[p];
                    const int   ii = sidx[i], ip = sidx[p];
                    const bool  g  = (si > sp) || (si == sp && ii < ip);
                    if (g != dir) {
                        s[i] = sp; s[p] = si;
                        sidx[i] = ip; sidx[p] = ii;
                    }
                }
            }
            __syncthreads();
        }
    }

    if (tid < TOPK) out_scores[(size_t)b * TOPK + tid] = s[tid];

    // gather 64 winning rows of 769 floats each (coalesced row copies)
    for (int j = 0; j < TOPK; ++j) {
        const int src = sidx[j];
        const float* srow = cand + ((size_t)b * Kn + src) * Dn;
        float*       drow = hard + ((size_t)b * TOPK + j) * Dn;
        for (int d = tid; d < Dn; d += 256) drow[d] = srow[d];
    }
}

extern "C" void kernel_launch(void* const* d_in, const int* in_sizes, int n_in,
                              void* d_out, int out_size, void* d_ws, size_t ws_size,
                              hipStream_t stream)
{
    const float* anchor = (const float*)d_in[0];   // [B,E]
    const float* neg    = (const float*)d_in[1];   // [B,K,E]
    const float* cand   = (const float*)d_in[2];   // [B,K,D]
    // d_in[3] is k (==64), folded into TOPK at compile time.

    float* scores     = (float*)d_ws;                      // B*K floats
    float* hard       = (float*)d_out;                     // B*TOPK*D
    float* out_scores = hard + (size_t)Bn * TOPK * Dn;     // B*TOPK

    rg_score_kernel<<<Bn, 256, 0, stream>>>(anchor, neg, scores);
    rg_topk_gather_kernel<<<Bn, 256, 0, stream>>>(scores, cand, hard, out_scores);
}
